// Criterion_35012573397697
// MI455X (gfx1250) — matrix operand order repaired and checked by reading, not compiled, not run on hardware
//
#include <hip/hip_runtime.h>
#include <math.h>

#define DK   128          // feature dim (reference: 128)
#define LDP  (DK + 8)     // LDS row pitch in halves: 272B row stride -> 4-bank offset, conflict-free
#define BM   64           // batch rows per workgroup
#define BN   64           // proxy cols per inner tile
#define NITER 16          // column tiles per workgroup

typedef __attribute__((ext_vector_type(16))) _Float16 v16h;
typedef __attribute__((ext_vector_type(8)))  _Float16 v8h;
typedef __attribute__((ext_vector_type(4)))  _Float16 v4h;
typedef __attribute__((ext_vector_type(8)))  float    v8f;

__device__ __forceinline__ float wave_sum32(float v) {
  v += __shfl_xor(v, 16);
  v += __shfl_xor(v, 8);
  v += __shfl_xor(v, 4);
  v += __shfl_xor(v, 2);
  v += __shfl_xor(v, 1);
  return v;
}

// Async copy of 64 contiguous bytes (4 x b128) from global to LDS, tracked by ASYNCcnt.
// VGLOBAL async form: vdst = LDS byte offset VGPR, vaddr = 32-bit per-lane byte offset,
// saddr = 64-bit uniform base; INST_OFFSET is added to BOTH the LDS and global addresses.
__device__ __forceinline__ void async_copy64(unsigned int lds_off, unsigned int voff,
                                             const void* sbase) {
  asm volatile(
      "global_load_async_to_lds_b128 %0, %1, %2 offset:0\n\t"
      "global_load_async_to_lds_b128 %0, %1, %2 offset:16\n\t"
      "global_load_async_to_lds_b128 %0, %1, %2 offset:32\n\t"
      "global_load_async_to_lds_b128 %0, %1, %2 offset:48"
      :: "v"(lds_off), "v"(voff), "s"(sbase) : "memory");
}

// One wave32 per row: L2-normalize*3 -> f16, write row squared norm, zero rowsum.
__global__ void k_prep_batch(const float* __restrict__ src, _Float16* __restrict__ dst,
                             float* __restrict__ sq, float* __restrict__ rowsum, int nrows) {
  const int wave = threadIdx.x >> 5, lane = threadIdx.x & 31;
  const int row = blockIdx.x * 8 + wave;
  if (row >= nrows) return;
  const float4 v = *(const float4*)(src + (size_t)row * DK + lane * 4);
  float s = v.x * v.x + v.y * v.y + v.z * v.z + v.w * v.w;
  s = wave_sum32(s);
  const float sc = 3.0f / fmaxf(sqrtf(s), 1e-12f);
  v4h h;
  h[0] = (_Float16)(v.x * sc); h[1] = (_Float16)(v.y * sc);
  h[2] = (_Float16)(v.z * sc); h[3] = (_Float16)(v.w * sc);
  *(v4h*)(dst + (size_t)row * DK + lane * 4) = h;
  if (lane == 0) { sq[row] = s * sc * sc; rowsum[row] = 0.0f; }
}

// One wave32 per proxy row; rows >= C are padding: zero data, huge p2 so exp(-dist)=0.
__global__ void k_prep_prox(const float* __restrict__ src, _Float16* __restrict__ dst,
                            float* __restrict__ sq, int C, int Cpad) {
  const int wave = threadIdx.x >> 5, lane = threadIdx.x & 31;
  const int row = blockIdx.x * 8 + wave;
  if (row >= Cpad) return;
  if (row >= C) {
    v4h z = {};
    *(v4h*)(dst + (size_t)row * DK + lane * 4) = z;
    if (lane == 0) sq[row] = 1e30f;
    return;
  }
  const float4 v = *(const float4*)(src + (size_t)row * DK + lane * 4);
  float s = v.x * v.x + v.y * v.y + v.z * v.z + v.w * v.w;
  s = wave_sum32(s);
  const float sc = 3.0f / fmaxf(sqrtf(s), 1e-12f);
  v4h h;
  h[0] = (_Float16)(v.x * sc); h[1] = (_Float16)(v.y * sc);
  h[2] = (_Float16)(v.z * sc); h[3] = (_Float16)(v.w * sc);
  *(v4h*)(dst + (size_t)row * DK + lane * 4) = h;
  if (lane == 0) sq[row] = s * sc * sc;
}

// Exact f32 positive-pair distance: one wave32 per batch row.
__global__ void k_pos(const float* __restrict__ xb, const float* __restrict__ pr,
                      const int* __restrict__ lab, float* __restrict__ pos, int B) {
  const int wave = threadIdx.x >> 5, lane = threadIdx.x & 31;
  const int b = blockIdx.x * 8 + wave;
  if (b >= B) return;
  const int c = lab[b];
  const float4 xv = *(const float4*)(xb + (size_t)b * DK + lane * 4);
  const float4 pv = *(const float4*)(pr + (size_t)c * DK + lane * 4);
  float sb = xv.x * xv.x + xv.y * xv.y + xv.z * xv.z + xv.w * xv.w;
  float sp = pv.x * pv.x + pv.y * pv.y + pv.z * pv.z + pv.w * pv.w;
  float dp = xv.x * pv.x + xv.y * pv.y + xv.z * pv.z + xv.w * pv.w;
  sb = wave_sum32(sb); sp = wave_sum32(sp); dp = wave_sum32(dp);
  const float sx = 3.0f / fmaxf(sqrtf(sb), 1e-12f);
  const float sy = 3.0f / fmaxf(sqrtf(sp), 1e-12f);
  if (lane == 0) pos[b] = sb * sx * sx + sp * sy * sy - 2.0f * dp * sx * sy;
}

// Main fused GEMM + exp + per-row negative sum.
// Async double-buffered proxy tiles; A fragments hoisted to registers for all 16 tiles.
// 256 threads = 8 waves; each wave owns a 16-row strip and two 16-col tiles of the 64x64 block.
__global__ __launch_bounds__(256)
void k_main(const _Float16* __restrict__ xh, const _Float16* __restrict__ ph,
            const float* __restrict__ x2, const float* __restrict__ p2,
            const int* __restrict__ lab, float* __restrict__ rowsum, int ntiles) {
  __shared__ _Float16 xs[BM * LDP];
  __shared__ _Float16 ps[2][BN * LDP];

  const int tid  = threadIdx.x;
  const int lane = tid & 31, wave = tid >> 5;
  const int m0    = blockIdx.y * BM;
  const int tile0 = blockIdx.x * NITER;
  int count = ntiles - tile0;
  if (count > NITER) count = NITER;

  // Per-thread async-copy geometry: thread t feeds row r = t>>2, 32 halves at (t&3)*32.
  const int cprow = tid >> 2;
  const int cpseg = (tid & 3) * 32;
  const unsigned int gvoff = (unsigned int)((cprow * DK + cpseg) * 2);  // global byte offset
  const unsigned int xs_lds =
      (unsigned int)(size_t)(&xs[cprow * LDP + cpseg]);                 // LDS byte offset
  const unsigned int ps_lds0 =
      (unsigned int)(size_t)(&ps[0][cprow * LDP + cpseg]);
  const unsigned int ps_lds1 =
      (unsigned int)(size_t)(&ps[1][cprow * LDP + cpseg]);

  // Prologue: async-fetch x tile and proxy tile 0 (both covered by first wait+barrier).
  async_copy64(xs_lds, gvoff, (const void*)(xh + (size_t)m0 * DK));
  async_copy64(ps_lds0, gvoff, (const void*)(ph + (size_t)(tile0 * BN) * DK));

  const int tr  = wave & 3;          // 16-row strip within 64
  const int tc0 = (wave >> 2) * 2;   // first of two 16-col tiles
  const int rbase = m0 + 16 * tr + 8 * (lane >> 4);  // D-layout: lane<16 -> M=r, lane>=16 -> M=r+8

  float px2[8]; int plab[8]; float psum[8];
#pragma unroll
  for (int r = 0; r < 8; ++r) {
    px2[r]  = x2[rbase + r];
    plab[r] = lab[rbase + r];
    psum[r] = 0.0f;
  }

  // xs and ps[0] ready after this wait+barrier.
  asm volatile("s_wait_asynccnt 0x0" ::: "memory");
  __syncthreads();

  // Hoist the four A fragments (16x32 f16 each) into registers: invariant over all tiles.
  // Layout: lane holds row M=lane&15; halves 0..7 -> K=kb..kb+7, halves 8..15 -> K=kb+16..kb+23.
  v16h afrag[4];
#pragma unroll
  for (int kk = 0; kk < DK; kk += 32) {
    const _Float16* p = xs + (16 * tr + (lane & 15)) * LDP + kk + 8 * (lane >> 4);
    ((v8h*)&afrag[kk / 32])[0] = *(const v8h*)p;        // ds_load_b128
    ((v8h*)&afrag[kk / 32])[1] = *(const v8h*)(p + 16); // ds_load_b128
  }

  for (int it = 0; it < count; ++it) {
    if (it > 0) {
      // Tile `it` (issued last iteration) has landed when this wave's ASYNCcnt drains.
      asm volatile("s_wait_asynccnt 0x0" ::: "memory");
      __syncthreads();  // all waves' async data visible; all done reading the other buffer
    }

    // Kick off tile it+1 into the other buffer; overlaps with the compute below.
    if (it + 1 < count) {
      const int cn = (tile0 + it + 1) * BN;
      async_copy64(((it + 1) & 1) ? ps_lds1 : ps_lds0, gvoff,
                   (const void*)(ph + (size_t)cn * DK));
    }

    const _Float16* psbuf = ps[it & 1];
    const int c0 = (tile0 + it) * BN;

    v8f acc0 = {}; v8f acc1 = {};
#pragma unroll
    for (int kk = 0; kk < DK; kk += 32) {
      // B fragment 32x16 f16: lane holds col N=lane&15; same K layout as A fragment
      v16h b0, b1;
      {
        const _Float16* p = psbuf + (16 * tc0 + (lane & 15)) * LDP + kk + 8 * (lane >> 4);
        ((v8h*)&b0)[0] = *(const v8h*)p;
        ((v8h*)&b0)[1] = *(const v8h*)(p + 16);
        const _Float16* q = p + 16 * LDP;
        ((v8h*)&b1)[0] = *(const v8h*)q;
        ((v8h*)&b1)[1] = *(const v8h*)(q + 16);
      }
      acc0 = __builtin_amdgcn_wmma_f32_16x16x32_f16(false, afrag[kk / 32], false, b0,
                                                    (short)0, acc0, false, false);
      acc1 = __builtin_amdgcn_wmma_f32_16x16x32_f16(false, afrag[kk / 32], false, b1,
                                                    (short)0, acc1, false, false);
    }

    // Epilogue: dist = x2 + p2 - 2*dot; accumulate exp(-dist) over negatives.
    const int cg0 = c0 + 16 * tc0 + (lane & 15);
    const int cg1 = cg0 + 16;
    const float q0 = p2[cg0], q1 = p2[cg1];
#pragma unroll
    for (int r = 0; r < 8; ++r) {
      const float d0 = px2[r] + q0 - 2.0f * acc0[r];
      const float d1 = px2[r] + q1 - 2.0f * acc1[r];
      psum[r] += (plab[r] == cg0) ? 0.0f : __expf(-d0);
      psum[r] += (plab[r] == cg1) ? 0.0f : __expf(-d1);
    }
  }

  // Reduce across the 16 lanes that share each output row, then one atomic per row.
#pragma unroll
  for (int r = 0; r < 8; ++r) {
    float s = psum[r];
    s += __shfl_xor(s, 1);
    s += __shfl_xor(s, 2);
    s += __shfl_xor(s, 4);
    s += __shfl_xor(s, 8);
    if ((lane & 15) == 0) atomicAdd(&rowsum[rbase + r], s);
  }
}

__global__ void k_final(const float* __restrict__ pos, const float* __restrict__ rowsum,
                        float* __restrict__ out, int B) {
  __shared__ float sm[256];
  float s = 0.0f;
  for (int b = threadIdx.x; b < B; b += 256) s += pos[b] + logf(rowsum[b]);
  sm[threadIdx.x] = s;
  __syncthreads();
  for (int st = 128; st > 0; st >>= 1) {
    if (threadIdx.x < st) sm[threadIdx.x] += sm[threadIdx.x + st];
    __syncthreads();
  }
  if (threadIdx.x == 0) out[0] = sm[0] / (float)B;
}

extern "C" void kernel_launch(void* const* d_in, const int* in_sizes, int n_in,
                              void* d_out, int out_size, void* d_ws, size_t ws_size,
                              hipStream_t stream) {
  (void)n_in; (void)out_size; (void)ws_size;
  const float* batch   = (const float*)d_in[0];
  const int*   labels  = (const int*)d_in[1];
  const float* proxies = (const float*)d_in[2];

  const int B = in_sizes[1];                 // 512
  const int D = in_sizes[0] / B;             // 128
  const int C = in_sizes[2] / D;             // 100000
  const int Cpad = (C + 63) & ~63;           // 100032

  char* ws = (char*)d_ws;
  size_t off = 0;
  auto take = [&](size_t bytes) -> char* {
    char* p = ws + off;
    off = (off + bytes + 255) & ~(size_t)255;
    return p;
  };
  _Float16* xh     = (_Float16*)take((size_t)B * D * sizeof(_Float16));
  _Float16* ph     = (_Float16*)take((size_t)Cpad * D * sizeof(_Float16));
  float*    x2     = (float*)take((size_t)B * sizeof(float));
  float*    p2     = (float*)take((size_t)Cpad * sizeof(float));
  float*    rowsum = (float*)take((size_t)B * sizeof(float));
  float*    pos    = (float*)take((size_t)B * sizeof(float));

  k_prep_batch<<<(B + 7) / 8, 256, 0, stream>>>(batch, xh, x2, rowsum, B);
  k_prep_prox<<<(Cpad + 7) / 8, 256, 0, stream>>>(proxies, ph, p2, C, Cpad);
  k_pos<<<(B + 7) / 8, 256, 0, stream>>>(batch, proxies, labels, pos, B);

  const int ntiles = Cpad / BN;              // 1563 column tiles
  dim3 grid((ntiles + NITER - 1) / NITER, B / BM);  // (98, 8)
  k_main<<<grid, 256, 0, stream>>>(xh, ph, x2, p2, labels, rowsum, ntiles);

  k_final<<<1, 256, 0, stream>>>(pos, rowsum, (float*)d_out, B);
}
